// LSTMnet_59725815218259
// MI455X (gfx1250) — matrix operand order repaired and checked by reading, not compiled, not run on hardware
//
#include <hip/hip_runtime.h>
#include <hip/hip_bf16.h>

typedef __attribute__((ext_vector_type(16))) _Float16 v16h;
typedef __attribute__((ext_vector_type(8)))  _Float16 v8h;
typedef __attribute__((ext_vector_type(4)))  _Float16 v4h;
typedef __attribute__((ext_vector_type(8)))  float    v8f;
typedef __attribute__((ext_vector_type(4)))  float    v4f;

#define T_STEPS   128
#define HID       64
#define NGATES    256
#define TILE_M    16
#define NTHREADS  256
#define N_SEQ     207   // N
#define B_BATCH   64    // B
#define F_IN      2

// CDNA5 has a hardware TANH transcendental (TRANS32 class). Use it when the
// builtin is available; otherwise fall back to v_exp + v_rcp (still no IEEE
// divide on the serial recurrence path).
__device__ __forceinline__ float fast_tanh(float x) {
#if __has_builtin(__builtin_amdgcn_tanhf)
    return __builtin_amdgcn_tanhf(x);
#else
    const float e = __expf(2.0f * x);
    return 1.0f - 2.0f * __builtin_amdgcn_rcpf(e + 1.0f);
#endif
}

__device__ __forceinline__ float sigmoid_f(float x) {
    // sigmoid(x) = 0.5 + 0.5*tanh(0.5*x): one TRANS op + 2 VALU, no divide.
    return __builtin_fmaf(0.5f, fast_tanh(0.5f * x), 0.5f);
}

// A-fragment (16x32 f16) for v_wmma_f32_16x16x32_f16, loaded from an LDS
// row-major [16][64] f16 buffer. row_base = &buf[r][ks*32].
// Element j of the per-lane v16h holds k = half*16 + j.
// j=0..7  -> k = half*16 .. half*16+7      (contiguous, one b128 load)
// j=8..15 -> k = half*16+8 .. +7           (contiguous, one b128 load)
__device__ __forceinline__ v16h load_a_frag(const _Float16* row_base, int half) {
    const v8h lo = *(const v8h*)(row_base + half * 16);
    const v8h hi = *(const v8h*)(row_base + half * 16 + 8);
    return __builtin_shufflevector(lo, hi, 0, 1, 2, 3, 4, 5, 6, 7,
                                           8, 9, 10, 11, 12, 13, 14, 15);
}

// B-fragment (32x16 f16): lane holds column n = r, element j holds
// k = half*16 + j. For gates = h @ W^T we need B[k][n] = W[gcol+n][k],
// so per-lane this is 16 *contiguous* fp32 values of one W row:
// p = W + (gcol + r)*64 + ks*32 + half*16. Converted fp32 -> f16 once.
__device__ __forceinline__ v16h load_w_frag(const float* p) {
    v16h o;
#pragma unroll
    for (int j = 0; j < 16; ++j) o[j] = (_Float16)p[j];
    return o;
}

__device__ __forceinline__ v8f wmma16x16x32(v16h a, v16h b, v8f c) {
    return __builtin_amdgcn_wmma_f32_16x16x32_f16(
        /*neg_a=*/false, a, /*neg_b=*/false, b,
        /*c_mod=*/(short)0, c, /*reuse_a=*/false, /*reuse_b=*/false);
}

__global__ __launch_bounds__(NTHREADS)
void lstm2_fused_kernel(const float* __restrict__ x,
                        const float* __restrict__ Wih0,   // [256][2]
                        const float* __restrict__ Whh0,   // [256][64]
                        const float* __restrict__ bih0,   // [256]
                        const float* __restrict__ bhh0,   // [256]
                        const float* __restrict__ Wih1,   // [256][64]
                        const float* __restrict__ Whh1,   // [256][64]
                        const float* __restrict__ bih1,   // [256]
                        const float* __restrict__ bhh1,   // [256]
                        const float* __restrict__ W1,     // [128][64]
                        const float* __restrict__ b1,     // [128]
                        const float* __restrict__ W2,     // [2][128]
                        const float* __restrict__ b2,     // [2]
                        float* __restrict__ out)          // [B][2][1][N]
{
    __shared__ __align__(16) float    xt[T_STEPS][TILE_M][F_IN];   // 16 KB
    __shared__ __align__(16) float    gls[TILE_M][NGATES];         // 16 KB
    __shared__ __align__(16) _Float16 hb0[TILE_M][HID];            // 2 KB
    __shared__ __align__(16) _Float16 hb1[TILE_M][HID];            // 2 KB
    __shared__ __align__(16) float    bsum0[NGATES];
    __shared__ __align__(16) float    bsum1[NGATES];
    __shared__ __align__(16) float    wih0s[NGATES][2];
    __shared__ __align__(16) float    b1s[128];
    __shared__ __align__(16) float    w2s[2][128];
    __shared__ __align__(16) float    b2s[2];
    __shared__ __align__(16) _Float16 ybuf[TILE_M][128];           // 4 KB

    const int tid  = threadIdx.x;
    const int wave = tid >> 5;
    const int lane = tid & 31;
    const int half = lane >> 4;
    const int r    = lane & 15;
    const int tile = blockIdx.x;           // 828 tiles of 16 rows over M=13248

    // Elementwise work assignment: thread owns 4 consecutive h at fixed m.
    const int em = tid >> 4;               // row m (0..15)
    const int eh = (tid & 15) * 4;         // first of 4 consecutive h

    // ---- Stage x tile + biases into LDS (one time) ----
    {
        float* xflat = &xt[0][0][0];
        for (int i = tid; i < T_STEPS * TILE_M * F_IN; i += NTHREADS) {
            int f  = i & 1;
            int m  = (i >> 1) & (TILE_M - 1);
            int t  = i >> 5;
            int gm = tile * TILE_M + m;
            int b  = gm / N_SEQ;
            int n  = gm - b * N_SEQ;
            // x is (B, F, T, N)
            xflat[i] = x[((b * F_IN + f) * T_STEPS + t) * N_SEQ + n];
        }
        for (int i = tid; i < NGATES; i += NTHREADS) {
            bsum0[i]    = bih0[i] + bhh0[i];
            bsum1[i]    = bih1[i] + bhh1[i];
            wih0s[i][0] = Wih0[i * 2 + 0];
            wih0s[i][1] = Wih0[i * 2 + 1];
        }
        for (int i = tid; i < 128; i += NTHREADS) {
            b1s[i]    = b1[i];
            w2s[0][i] = W2[i];
            w2s[1][i] = W2[128 + i];
        }
        if (tid < 2) b2s[tid] = b2[tid];
        _Float16* h0f = &hb0[0][0];
        _Float16* h1f = &hb1[0][0];
        for (int i = tid; i < TILE_M * HID; i += NTHREADS) {
            h0f[i] = (_Float16)0.0f;
            h1f[i] = (_Float16)0.0f;
        }
    }

    // ---- Pre-pack weight B-fragments into registers (reused 128x) ----
    // Wave w owns gate columns [w*32, w*32+32): 2 N-tiles x 2 K-steps each.
    v16h Bhh0[2][2], Bih1[2][2], Bhh1[2][2];
#pragma unroll
    for (int nt = 0; nt < 2; ++nt) {
        const int g = wave * 32 + nt * 16 + r;
#pragma unroll
        for (int ks = 0; ks < 2; ++ks) {
            const int ko = ks * 32 + half * 16;
            Bhh0[nt][ks] = load_w_frag(Whh0 + g * HID + ko);
            Bih1[nt][ks] = load_w_frag(Wih1 + g * HID + ko);
            Bhh1[nt][ks] = load_w_frag(Whh1 + g * HID + ko);
        }
    }

    // Cell state lives in registers; thread owns (em, eh..eh+3) pairs.
    float c0[4] = {0.f, 0.f, 0.f, 0.f};
    float c1[4] = {0.f, 0.f, 0.f, 0.f};

    __syncthreads();

    for (int t = 0; t < T_STEPS; ++t) {
        // h2_{t-1} fragments: hb1 is not rewritten until the end of this
        // step, so issue these early to overlap the whole layer-0 phase.
        v16h p0 = load_a_frag(&hb1[r][0], half);
        v16h p1 = load_a_frag(&hb1[r][32], half);

        // ================= Layer 0: gates = h1_{t-1} @ Whh0^T + x@Wih0^T + b
        {
            v16h a0 = load_a_frag(&hb0[r][0], half);
            v16h a1 = load_a_frag(&hb0[r][32], half);
#pragma unroll
            for (int nt = 0; nt < 2; ++nt) {
                v8f acc = {};
                acc = wmma16x16x32(a0, Bhh0[nt][0], acc);
                acc = wmma16x16x32(a1, Bhh0[nt][1], acc);
                const int g  = wave * 32 + nt * 16 + r;
                const float bs = bsum0[g];
                const float w0 = wih0s[g][0];
                const float w1 = wih0s[g][1];
#pragma unroll
                for (int v = 0; v < 8; ++v) {
                    const int m = v + 8 * half;       // C layout: row = v + 8*half
                    gls[m][g] = acc[v] + bs + xt[t][m][0] * w0 + xt[t][m][1] * w1;
                }
            }
        }
        __syncthreads();

        // elementwise layer 0 -> h1_t (f16) into hb0 (vectorized LDS access)
        {
            const v4f iv = *(const v4f*)&gls[em][eh];
            const v4f fv = *(const v4f*)&gls[em][64 + eh];
            const v4f gv = *(const v4f*)&gls[em][128 + eh];
            const v4f ov = *(const v4f*)&gls[em][192 + eh];
            v4h hv;
#pragma unroll
            for (int p = 0; p < 4; ++p) {
                const float ig = sigmoid_f(iv[p]);
                const float fg = sigmoid_f(fv[p]);
                const float gg = fast_tanh(gv[p]);
                const float og = sigmoid_f(ov[p]);
                c0[p] = fg * c0[p] + ig * gg;
                hv[p] = (_Float16)(og * fast_tanh(c0[p]));
            }
            *(v4h*)&hb0[em][eh] = hv;
        }
        __syncthreads();

        // ======= Layer 1: gates = h1_t @ Wih1^T + h2_{t-1} @ Whh1^T + b
        {
            v16h a0 = load_a_frag(&hb0[r][0], half);
            v16h a1 = load_a_frag(&hb0[r][32], half);
#pragma unroll
            for (int nt = 0; nt < 2; ++nt) {
                v8f acc = {};
                acc = wmma16x16x32(a0, Bih1[nt][0], acc);
                acc = wmma16x16x32(a1, Bih1[nt][1], acc);
                acc = wmma16x16x32(p0, Bhh1[nt][0], acc);
                acc = wmma16x16x32(p1, Bhh1[nt][1], acc);
                const int g = wave * 32 + nt * 16 + r;
                const float bs = bsum1[g];
#pragma unroll
                for (int v = 0; v < 8; ++v) {
                    const int m = v + 8 * half;
                    gls[m][g] = acc[v] + bs;
                }
            }
        }
        __syncthreads();

        // elementwise layer 1 -> h2_t (f16) into hb1
        {
            const v4f iv = *(const v4f*)&gls[em][eh];
            const v4f fv = *(const v4f*)&gls[em][64 + eh];
            const v4f gv = *(const v4f*)&gls[em][128 + eh];
            const v4f ov = *(const v4f*)&gls[em][192 + eh];
            v4h hv;
#pragma unroll
            for (int p = 0; p < 4; ++p) {
                const float ig = sigmoid_f(iv[p]);
                const float fg = sigmoid_f(fv[p]);
                const float gg = fast_tanh(gv[p]);
                const float og = sigmoid_f(ov[p]);
                c1[p] = fg * c1[p] + ig * gg;
                hv[p] = (_Float16)(og * fast_tanh(c1[p]));
            }
            *(v4h*)&hb1[em][eh] = hv;
        }
        __syncthreads();
    }

    // ================= Head: y1 = relu(h2 @ W1^T + b1), y2 = y1 @ W2^T + b2
    {
        v16h a0 = load_a_frag(&hb1[r][0], half);
        v16h a1 = load_a_frag(&hb1[r][32], half);
        const int col = wave * 16;                   // 8 waves x 16 cols = 128
        v16h bw0 = load_w_frag(W1 + (col + r) * HID + half * 16);
        v16h bw1 = load_w_frag(W1 + (col + r) * HID + 32 + half * 16);
        v8f acc = {};
        acc = wmma16x16x32(a0, bw0, acc);
        acc = wmma16x16x32(a1, bw1, acc);
        const float bb = b1s[col + r];
#pragma unroll
        for (int v = 0; v < 8; ++v) {
            float y = acc[v] + bb;
            y = y > 0.0f ? y : 0.0f;
            ybuf[v + 8 * half][col + r] = (_Float16)y;
        }
    }
    __syncthreads();

    if (tid < 32) {
        const int m = tid >> 1;
        const int c = tid & 1;
        float s = b2s[c];
#pragma unroll 8
        for (int k = 0; k < 128; ++k)
            s += (float)ybuf[m][k] * w2s[c][k];
        const int gm = tile * TILE_M + m;
        const int b  = gm / N_SEQ;
        const int n  = gm - b * N_SEQ;
        // out is (B, 2, 1, N)
        out[(b * 2 + c) * N_SEQ + n] = s;
    }
}

extern "C" void kernel_launch(void* const* d_in, const int* in_sizes, int n_in,
                              void* d_out, int out_size, void* d_ws, size_t ws_size,
                              hipStream_t stream) {
    (void)in_sizes; (void)n_in; (void)d_ws; (void)ws_size; (void)out_size;
    const float* x    = (const float*)d_in[0];
    const float* Wih0 = (const float*)d_in[1];
    const float* Whh0 = (const float*)d_in[2];
    const float* bih0 = (const float*)d_in[3];
    const float* bhh0 = (const float*)d_in[4];
    const float* Wih1 = (const float*)d_in[5];
    const float* Whh1 = (const float*)d_in[6];
    const float* bih1 = (const float*)d_in[7];
    const float* bhh1 = (const float*)d_in[8];
    const float* W1   = (const float*)d_in[9];
    const float* b1   = (const float*)d_in[10];
    const float* W2   = (const float*)d_in[11];
    const float* b2   = (const float*)d_in[12];
    float* out = (float*)d_out;

    const int M = B_BATCH * N_SEQ;           // 13248
    const int nblocks = M / TILE_M;          // 828 (exact)
    lstm2_fused_kernel<<<nblocks, NTHREADS, 0, stream>>>(
        x, Wih0, Whh0, bih0, bhh0, Wih1, Whh1, bih1, bhh1,
        W1, b1, W2, b2, out);
}
